// SPEBiasedMultiHeadAttention_51393578664577
// MI455X (gfx1250) — matrix operand order repaired and checked by reading, not compiled, not run on hardware
//
#include <hip/hip_runtime.h>
#include <hip/hip_bf16.h>

// ---- CDNA5 WMMA / TDM types (gfx1250, wave32) ----
typedef __attribute__((ext_vector_type(16))) __bf16 v16bf;
typedef __attribute__((ext_vector_type(8)))  __bf16 v8bf;
typedef __attribute__((ext_vector_type(8)))  float  v8f;
typedef __attribute__((ext_vector_type(4)))  unsigned int v4u;
typedef __attribute__((ext_vector_type(4)))  int    v4i;
typedef __attribute__((ext_vector_type(8)))  int    v8i;

union BF16x16 { v16bf v; v8bf h2[2]; __bf16 e[16]; };
union F32x8   { v8f v; float e[8]; };

// Problem constants
#define BT_N   512
#define E_LEN  256
#define DMODEL 128
#define NH     8
#define DH     16

// ---------------------------------------------------------------------------
// Kernel 1: convert the 4 weight matrices (128x128 fp32) to bf16 in workspace
// ---------------------------------------------------------------------------
__global__ void k_wconv(const float* __restrict__ Wq, const float* __restrict__ Wk,
                        const float* __restrict__ Wv, const float* __restrict__ Wo,
                        __bf16* __restrict__ Wb) {
    int idx = blockIdx.x * 256 + threadIdx.x;              // 65536 total
    const float* src = (idx < 16384) ? Wq : (idx < 32768) ? Wk
                     : (idx < 49152) ? Wv : Wo;
    Wb[idx] = (__bf16)src[idx & 16383];
}

// ---------------------------------------------------------------------------
// Kernel 2: bias = (spe @ spe^T) / sqrt(16)   -> (256,256) fp32
// ---------------------------------------------------------------------------
__global__ void k_bias(const float* __restrict__ spe, float* __restrict__ bias) {
    int i = blockIdx.x, j = threadIdx.x;
    float s = 0.f;
#pragma unroll
    for (int p = 0; p < 16; ++p) s += spe[i * 16 + p] * spe[j * 16 + p];
    bias[i * 256 + j] = s * 0.25f;
}

// ---------------------------------------------------------------------------
// Kernel 3: fused Q/K/V projection.
// Block = 128 threads (4 waves). Each block owns a 16x128 row-tile of x
// (rows are flattened b*E+e), stages it as bf16 in LDS, then each wave runs
// WMMA output tiles for all 3 weight matrices (24 tiles, K-loop of 4).
// Output layout: Q/K/V[row = b*E+e][col = head*16 + d]  (bf16).
// ---------------------------------------------------------------------------
__global__ void k_proj3(const float* __restrict__ x, const __bf16* __restrict__ Wb,
                        __bf16* __restrict__ Q, __bf16* __restrict__ K,
                        __bf16* __restrict__ V) {
    __shared__ __align__(16) __bf16 sA[16 * 128];
    const int tid = threadIdx.x;
    const size_t rowBase = (size_t)blockIdx.x * 16;

    for (int i = tid; i < 16 * 128; i += 128)              // fp32 -> bf16 stage
        sA[i] = (__bf16)x[rowBase * 128 + i];
    __syncthreads();

    const int lane = tid & 31, wave = tid >> 5;
    const int m = lane & 15, h = lane >> 4;                // A: row, K-half select

    // A fragments for all 4 K-steps (ISA 16-bit A layout: elems 0..7 -> K=h*8+i,
    // elems 8..15 -> K=16+h*8+(i-8)); both chunks are 16B-contiguous in LDS.
    BF16x16 a[4];
#pragma unroll
    for (int kk = 0; kk < 4; ++kk) {
        a[kk].h2[0] = *(const v8bf*)&sA[m * 128 + kk * 32 + h * 8];
        a[kk].h2[1] = *(const v8bf*)&sA[m * 128 + kk * 32 + 16 + h * 8];
    }

    for (int t = wave; t < 24; t += 4) {                   // 3 matrices x 8 N-tiles
        const int mat = t >> 3, n = t & 7;
        const __bf16* W = Wb + mat * 16384;
        F32x8 c = {};
#pragma unroll
        for (int kk = 0; kk < 4; ++kk) {
            // B[k][n] = W[n][k]: per-lane 2 x 8 contiguous bf16 (k=h*8+i, 16+h*8+i) from W row (n*16+m)
            BF16x16 bv;
            const __bf16* bp = W + (size_t)(n * 16 + m) * 128 + kk * 32 + h * 8;
            bv.h2[0] = *(const v8bf*)bp;
            bv.h2[1] = *(const v8bf*)(bp + 16);
            c.v = __builtin_amdgcn_wmma_f32_16x16x32_bf16(
                false, a[kk].v, false, bv.v, (short)0, c.v, false, false);
        }
        __bf16* outp = (mat == 0) ? Q : (mat == 1) ? K : V;
#pragma unroll
        for (int r = 0; r < 8; ++r)                        // C layout: m=r+8h, n=lane%16
            outp[(rowBase + r + 8 * h) * 128 + n * 16 + m] = (__bf16)c.e[r];
    }
}

// ---------------------------------------------------------------------------
// Kernel 4: attention core. One wave per (b, head, q-tile of 16).
//  - At wave start, a Tensor Data Mover descriptor DMAs the (256 keys x 16 dh)
//    bf16 V tile for this (b,head) into LDS (overlaps QK^T + softmax).
//  - 16 WMMAs -> full 16x256 logits in registers; softmax via 16-lane
//    shfl_xor reductions; fp32 attn_weights written out.
//  - s_wait_tensorcnt 0, then 8 WMMAs for P@V with B-frags gathered from LDS.
// ---------------------------------------------------------------------------
__global__ void k_attn(const __bf16* __restrict__ Q, const __bf16* __restrict__ K,
                       const __bf16* __restrict__ V, const float* __restrict__ bias,
                       float* __restrict__ attnw, __bf16* __restrict__ AO) {
    __shared__ __align__(16) __bf16 sP[16 * 256];          // probabilities (q x key)
    __shared__ __align__(16) __bf16 sV[256 * 16];          // V tile (key x dh), TDM dest
    const int bid = blockIdx.x;
    const int qt = bid & 15, hd = (bid >> 4) & 7, b = bid >> 7;
    const int lane = threadIdx.x, nl = lane & 15, h = lane >> 4;

    // ---- Issue TDM: 2-D tile (tile_dim0=16 elems/row, tile_dim1=256 rows) of
    // bf16 (data_size=1 -> 2B) from V[b*256 .. +255][hd*16 .. +15] into sV.
    // D# bit layout per CDNA5 ISA 8.3/8.4. Uniform operands -> SGPR descriptor.
    {
        unsigned long long ga =
            (unsigned long long)(uintptr_t)&V[((size_t)b * 256) * 128 + hd * 16];
        unsigned int ldsoff = (unsigned int)(uintptr_t)(void*)sV;   // LDS aperture low bits
        v4u g0 = { 1u,                                  // count=1, user descriptor
                   ldsoff,                              // lds_addr  [63:32]
                   (unsigned int)ga,                    // global_addr[31:0]
                   (unsigned int)((ga >> 32) & 0x01FFFFFFu) | (2u << 30) }; // [56:32] | type=2
        v8i g1 = { (int)0x00010000,                     // wg_mask=0, data_size=1 (2B)
                   (int)(128u << 16),                   // tensor_dim0 = 128
                   (int)(256u << 16),                   // tensor_dim1 = 256
                   (int)(16u  << 16),                   // tile_dim0  = 16
                   (int)256,                            // tile_dim1  = 256, tile_dim2 = 0
                   (int)128,                            // tensor_dim0_stride = 128
                   0, 0 };
        v4i z4 = { 0, 0, 0, 0 };
#if defined(__clang_major__) && (__clang_major__ >= 23)
        v8i z8 = { 0, 0, 0, 0, 0, 0, 0, 0 };
        __builtin_amdgcn_tensor_load_to_lds(g0, g1, z4, z4, z8, 0);
#else
        __builtin_amdgcn_tensor_load_to_lds(g0, g1, z4, z4, 0);
#endif
    }

    // ---- A fragment = Q tile, dh=16 zero-padded to K=32.
    BF16x16 a;
    a.h2[0] = *(const v8bf*)&Q[((size_t)b * 256 + qt * 16 + nl) * 128 + hd * 16 + h * 8];
#pragma unroll
    for (int i = 8; i < 16; ++i) a.e[i] = (__bf16)0.f;     // K=16..31 padding

    // ---- Logits: 16 key-tiles of 16x16 each
    F32x8 cc[16];
#pragma unroll
    for (int t = 0; t < 16; ++t) {
        BF16x16 bv;                                        // B[d][key] = Kmat[key][d]
        {                                                  // elems 0..7: d=h*8+i
            const __bf16* kp = &K[((size_t)b * 256 + t * 16 + nl) * 128 + hd * 16 + h * 8];
            bv.h2[0] = *(const v8bf*)kp;
        }                                                  // elems 8..15: d=16..31 padding
#pragma unroll
        for (int i = 8; i < 16; ++i) bv.e[i] = (__bf16)0.f;
        F32x8 c = {};
        c.v = __builtin_amdgcn_wmma_f32_16x16x32_bf16(
            false, a.v, false, bv.v, (short)0, c.v, false, false);
        cc[t] = c;
    }

    // ---- scale 1/sqrt(dh) + structural bias
#pragma unroll
    for (int t = 0; t < 16; ++t)
#pragma unroll
        for (int r = 0; r < 8; ++r)
            cc[t].e[r] = cc[t].e[r] * 0.25f
                       + bias[(qt * 16 + r + 8 * h) * 256 + t * 16 + nl];

    // ---- row softmax: each logit row lives across one 16-lane half (fixed r,h)
    float inv[8];
#pragma unroll
    for (int r = 0; r < 8; ++r) {
        float mx = -3.0e38f;
#pragma unroll
        for (int t = 0; t < 16; ++t) mx = fmaxf(mx, cc[t].e[r]);
#pragma unroll
        for (int d = 1; d < 16; d <<= 1) mx = fmaxf(mx, __shfl_xor(mx, d, 32));
        float s = 0.f;
#pragma unroll
        for (int t = 0; t < 16; ++t) {
            float p = __expf(cc[t].e[r] - mx);
            cc[t].e[r] = p;
            s += p;
        }
#pragma unroll
        for (int d = 1; d < 16; d <<= 1) s += __shfl_xor(s, d, 32);
        inv[r] = 1.0f / s;
    }

    // ---- write fp32 attention weights (b,h,q,k) + stage bf16 P in LDS (q x k)
    float* aw = attnw + (((size_t)(b * NH + hd) * 256) + qt * 16) * 256;
#pragma unroll
    for (int t = 0; t < 16; ++t)
#pragma unroll
        for (int r = 0; r < 8; ++r) {
            float p = cc[t].e[r] * inv[r];
            int ql = r + 8 * h;
            aw[(size_t)ql * 256 + t * 16 + nl] = p;
            sP[ql * 256 + t * 16 + nl] = (__bf16)p;
        }
    __syncthreads();

    // ---- wait for the TDM V tile, then P @ V over 256 keys in steps of 32
    __builtin_amdgcn_s_wait_tensorcnt(0);
    F32x8 acc = {};
#pragma unroll
    for (int kk = 0; kk < 8; ++kk) {
        BF16x16 ap, bv;
        ap.h2[0] = *(const v8bf*)&sP[nl * 256 + kk * 32 + h * 8];
        ap.h2[1] = *(const v8bf*)&sP[nl * 256 + kk * 32 + 16 + h * 8];
#pragma unroll
        for (int i = 0; i < 16; ++i) {                     // B[key][d] from LDS V tile
            int kr = kk * 32 + (i < 8 ? h * 8 + i : 16 + h * 8 + (i - 8));
            bv.e[i] = sV[kr * 16 + nl];
        }
        acc.v = __builtin_amdgcn_wmma_f32_16x16x32_bf16(
            false, ap.v, false, bv.v, (short)0, acc.v, false, false);
    }
#pragma unroll
    for (int r = 0; r < 8; ++r)                            // AO[b*E+q][hd*16+d] bf16
        AO[((size_t)b * 256 + qt * 16 + r + 8 * h) * 128 + hd * 16 + nl] = (__bf16)acc.e[r];
}

// ---------------------------------------------------------------------------
// Kernel 5: out = AO @ Wo^T  (bf16 WMMA, fp32 stores to d_out)
// ---------------------------------------------------------------------------
__global__ void k_outproj(const __bf16* __restrict__ AO, const __bf16* __restrict__ Wo,
                          float* __restrict__ out) {
    const int tid = threadIdx.x;                           // 128 threads, 4 waves
    const size_t rowBase = (size_t)blockIdx.x * 16;
    const int lane = tid & 31, wave = tid >> 5;
    const int m = lane & 15, h = lane >> 4;

    BF16x16 a[4];
#pragma unroll
    for (int kk = 0; kk < 4; ++kk) {
        const __bf16* ap = &AO[(rowBase + m) * 128];
        a[kk].h2[0] = *(const v8bf*)(ap + kk * 32 + h * 8);
        a[kk].h2[1] = *(const v8bf*)(ap + kk * 32 + 16 + h * 8);
    }

    for (int t = wave; t < 8; t += 4) {
        F32x8 c = {};
#pragma unroll
        for (int kk = 0; kk < 4; ++kk) {
            BF16x16 bv;
            const __bf16* bp = Wo + (size_t)(t * 16 + m) * 128 + kk * 32 + h * 8;
            bv.h2[0] = *(const v8bf*)bp;
            bv.h2[1] = *(const v8bf*)(bp + 16);
            c.v = __builtin_amdgcn_wmma_f32_16x16x32_bf16(
                false, a[kk].v, false, bv.v, (short)0, c.v, false, false);
        }
#pragma unroll
        for (int r = 0; r < 8; ++r)
            out[(rowBase + r + 8 * h) * 128 + t * 16 + m] = c.e[r];
    }
}

// ---------------------------------------------------------------------------
// Launch: all kernels on `stream` (graph-capture safe, no allocs/syncs).
// Workspace layout:
//   [0,128K)       : Wq|Wk|Wv|Wo bf16
//   [128K,384K)    : bias fp32 (256x256)
//   then Q, K, V, AO : 4 x 32MB bf16 (131072 rows x 128)
// d_out = out (512*256*128 fp32) ++ attn_weights (512*8*256*256 fp32)
// ---------------------------------------------------------------------------
extern "C" void kernel_launch(void* const* d_in, const int* in_sizes, int n_in,
                              void* d_out, int out_size, void* d_ws, size_t ws_size,
                              hipStream_t stream) {
    const float* x   = (const float*)d_in[0];
    const float* spe = (const float*)d_in[1];
    const float* Wq  = (const float*)d_in[2];
    const float* Wk  = (const float*)d_in[3];
    const float* Wv  = (const float*)d_in[4];
    const float* Wo  = (const float*)d_in[5];

    char* ws = (char*)d_ws;
    __bf16* Wb   = (__bf16*)ws;
    float*  bias = (float*)(ws + 131072);
    const size_t qkvBytes = (size_t)BT_N * E_LEN * DMODEL * sizeof(__bf16); // 32MB
    __bf16* Qb = (__bf16*)(ws + 393216);
    __bf16* Kb = (__bf16*)(ws + 393216 + qkvBytes);
    __bf16* Vb = (__bf16*)(ws + 393216 + 2 * qkvBytes);
    __bf16* AO = (__bf16*)(ws + 393216 + 3 * qkvBytes);

    float* out   = (float*)d_out;
    float* attnw = out + (size_t)BT_N * E_LEN * DMODEL;

    k_wconv<<<256, 256, 0, stream>>>(Wq, Wk, Wv, Wo, Wb);
    k_bias<<<256, 256, 0, stream>>>(spe, bias);
    k_proj3<<<(BT_N * E_LEN) / 16, 128, 0, stream>>>(x, Wb, Qb, Kb, Vb);
    k_attn<<<BT_N * NH * (E_LEN / 16), 32, 0, stream>>>(Qb, Kb, Vb, bias, attnw, AO);
    k_outproj<<<(BT_N * E_LEN) / 16, 128, 0, stream>>>(AO, Wb + 3 * 16384, out);
}
